// MultiHeadAttention_24189255811238
// MI455X (gfx1250) — matrix operand order repaired and checked by reading, not compiled, not run on hardware
//
#include <hip/hip_runtime.h>
#include <hip/hip_bf16.h>

typedef __attribute__((ext_vector_type(16))) _Float16 v16h;
typedef __attribute__((ext_vector_type(8)))  float    v8f;

#define T_SEQ 4096
#define C_DIM 768
#define NH    12
#define DH    64

union V16HU { v16h v; int4 q[2]; };

// A fragment: 16(M) x 32(K) f16 tile, row-major with stride S halves.
// ISA layout: lane<16 -> row=lane, K = {0..7, 16..23}; lane>=16 same rows, K = {8..15, 24..31}.
// Each lane's 16 halves are two contiguous 16B runs -> two ds_load_b128.
__device__ __forceinline__ v16h ld_frag_a(const _Float16* tile, int S) {
  const int lane = threadIdx.x & 31;
  const int row  = lane & 15;
  const int hl   = lane >> 4;
  const _Float16* p = tile + row * S + hl * 8;
  V16HU u;
  u.q[0] = *(const int4*)(p);
  u.q[1] = *(const int4*)(p + 16);
  return u.v;
}

// B fragment from a TRANSPOSED tile: tileT[n][k] (16 n-cols x 32 k), stride S halves.
// ISA layout: lane<16 -> col=lane, K = {0..7, 16..23}; lane>=16 -> K = {8..15, 24..31}.
__device__ __forceinline__ v16h ld_frag_bT(const _Float16* tileT, int S) {
  const int lane = threadIdx.x & 31;
  const int col  = lane & 15;
  const int hl   = lane >> 4;
  const _Float16* p = tileT + col * S + hl * 8;
  V16HU u;
  u.q[0] = *(const int4*)(p);
  u.q[1] = *(const int4*)(p + 16);
  return u.v;
}

// ---------------------------------------------------------------------------
// Kernel 1: qkv = x @ w_attn, fp32 -> f16, written as per-head Q/K/V buffers
// [H][T][Dh]. 64x64 output tile per 256-thread block; 8 waves x (16x32) tiles.
// K-chunk = 64: 4 WMMAs per wave per barrier round-trip.
// ---------------------------------------------------------------------------
__global__ __launch_bounds__(256)
void qkv_proj_wmma(const float* __restrict__ x, const float* __restrict__ w,
                   _Float16* __restrict__ qb, _Float16* __restrict__ kb,
                   _Float16* __restrict__ vb)
{
  constexpr int SA = 72;  // 64 + 8 pad halves (144B rows, 16B aligned)
  constexpr int SB = 72;
  __shared__ __align__(16) _Float16 lA [64 * SA];  // x tile  64(M) x 64(K)
  __shared__ __align__(16) _Float16 lBT[64 * SB];  // w tile stored [n][k], 64x64

  const int tid  = threadIdx.x;
  const int wv   = tid >> 5;
  const int lane = tid & 31;
  const int hl   = lane >> 4;
  const int nn   = lane & 15;
  const int row0 = blockIdx.x * 64;       // T tile
  const int col0 = blockIdx.y * 64;       // 3C tile
  const int rowg = (wv & 3) * 16;
  const int colg = (wv >> 2) * 32;

  v8f c0 = {}; v8f c1 = {};

  for (int k0 = 0; k0 < C_DIM; k0 += 64) {
    __syncthreads();
#pragma unroll
    for (int j = 0; j < 4; ++j) {
      const int id = tid * 4 + j;                       // 0..1023 float4s
      const int r  = id >> 4;                           // 0..63
      const int c4 = (id & 15) << 2;                    // 0..60
      {
        const float4 f = *(const float4*)(x + (size_t)(row0 + r) * C_DIM + k0 + c4);
        _Float16* d = &lA[r * SA + c4];
        d[0] = (_Float16)f.x; d[1] = (_Float16)f.y;
        d[2] = (_Float16)f.z; d[3] = (_Float16)f.w;
      }
      {
        const float4 f = *(const float4*)(w + (size_t)(k0 + r) * (3 * C_DIM) + col0 + c4);
        lBT[(c4 + 0) * SB + r] = (_Float16)f.x;
        lBT[(c4 + 1) * SB + r] = (_Float16)f.y;
        lBT[(c4 + 2) * SB + r] = (_Float16)f.z;
        lBT[(c4 + 3) * SB + r] = (_Float16)f.w;
      }
    }
    __syncthreads();

    const v16h a0  = ld_frag_a (&lA [rowg * SA],              SA);
    const v16h a1  = ld_frag_a (&lA [rowg * SA + 32],         SA);
    const v16h b00 = ld_frag_bT(&lBT[ colg       * SB],       SB);
    const v16h b01 = ld_frag_bT(&lBT[ colg       * SB + 32],  SB);
    const v16h b10 = ld_frag_bT(&lBT[(colg + 16) * SB],       SB);
    const v16h b11 = ld_frag_bT(&lBT[(colg + 16) * SB + 32],  SB);
    c0 = __builtin_amdgcn_wmma_f32_16x16x32_f16(false, a0, false, b00, (short)0, c0, false, false);
    c0 = __builtin_amdgcn_wmma_f32_16x16x32_f16(false, a1, false, b01, (short)0, c0, false, false);
    c1 = __builtin_amdgcn_wmma_f32_16x16x32_f16(false, a0, false, b10, (short)0, c1, false, false);
    c1 = __builtin_amdgcn_wmma_f32_16x16x32_f16(false, a1, false, b11, (short)0, c1, false, false);
  }

  // Tile -> {q,k,v} buffer + head is uniform per block (3C tiled by 64, 768/64 = 12 heads)
  const int tile  = col0 >> 6;            // 0..35
  const int which = tile / NH;            // 0=q 1=k 2=v
  const int head  = tile % NH;
  _Float16* dst = (which == 0) ? qb : (which == 1) ? kb : vb;
#pragma unroll
  for (int r = 0; r < 8; ++r) {
    const int grow = row0 + rowg + r + hl * 8;
    const size_t base = ((size_t)head * T_SEQ + grow) * DH;
    dst[base + colg + nn]      = (_Float16)c0[r];
    dst[base + colg + 16 + nn] = (_Float16)c1[r];
  }
}

// ---------------------------------------------------------------------------
// Kernel 2: causal flash attention. Block = (head, 128 query rows), 8 waves;
// each wave owns 16 full query rows. Streams 32-key K/V blocks via LDS.
// K tile uses the gfx1250 ASYNCcnt path (global_load_async_to_lds_b128);
// V goes through VGPRs because it is transposed in flight.
// scale = 768^-0.5 (faithful to the source, which uses the full model dim).
// ---------------------------------------------------------------------------
__global__ __launch_bounds__(256)
void flash_attn_wmma(const _Float16* __restrict__ qb, const _Float16* __restrict__ kb,
                     const _Float16* __restrict__ vb, float* __restrict__ out)
{
  constexpr int SK = 72;   // K tile [key][d], 144B rows
  constexpr int SV = 40;   // V tile stored transposed [d][key], 80B rows
  constexpr int SP = 40;   // per-wave P staging [m][key]
  __shared__ __align__(16) _Float16 lK [32 * SK];
  __shared__ __align__(16) _Float16 lVT[64 * SV];
  __shared__ __align__(16) _Float16 lP [8][16 * SP];

  const int tid  = threadIdx.x;
  const int wv   = tid >> 5;
  const int lane = tid & 31;
  const int hl   = lane >> 4;
  const int nn   = lane & 15;
  const int h    = blockIdx.y;
  const int q0   = blockIdx.x * 128;
  const int qrow_base = q0 + wv * 16;
  const size_t hbase  = (size_t)h * T_SEQ;

  // Preload this wave's Q rows as two A-fragments (d 0..31 and d 32..63).
  const _Float16* qp = qb + (hbase + qrow_base + nn) * DH;
  V16HU uq0, uq1;
  uq0.q[0] = *(const int4*)(qp + hl * 8);
  uq0.q[1] = *(const int4*)(qp + 16 + hl * 8);
  uq1.q[0] = *(const int4*)(qp + 32 + hl * 8);
  uq1.q[1] = *(const int4*)(qp + 48 + hl * 8);
  const v16h qf0 = uq0.v, qf1 = uq1.v;

  v8f o0 = {}, o1 = {}, o2 = {}, o3 = {};
  float Mrow[8], Lrow[8];
#pragma unroll
  for (int r = 0; r < 8; ++r) { Mrow[r] = -__builtin_inff(); Lrow[r] = 0.f; }

  const float scl = 0.03608439182435161f;      // 768^-0.5
  const int nblocks = (q0 >> 5) + 4;           // causal bound for this block
  const int key = tid >> 3;                    // cooperative K/V loader indices
  const int d0  = (tid & 7) << 3;
  // LDS byte offset of this lane's K-tile destination (generic ptr low 32 bits)
  const unsigned lK_off = (unsigned)(size_t)(&lK[key * SK + d0]);

  for (int ib = 0; ib < nblocks; ++ib) {
    const int kb0 = ib << 5;
    __syncthreads();
    {
      // K block: 32x64 f16 = 4KB = 256 lanes x 16B, async DMA into LDS
      const unsigned long long gk =
          (unsigned long long)(size_t)(kb + (hbase + kb0 + key) * DH + d0);
      asm volatile("global_load_async_to_lds_b128 %0, %1, off"
                   :: "v"(lK_off), "v"(gk) : "memory");

      // V block through VGPRs: transpose-store into lVT[d][key]
      const int4 vvec = *(const int4*)(vb + (hbase + kb0 + key) * DH + d0);
      const _Float16* vp = (const _Float16*)&vvec;
#pragma unroll
      for (int e = 0; e < 8; ++e) lVT[(d0 + e) * SV + key] = vp[e];

      if (ib + 1 < nblocks) {                  // global_prefetch_b8 next block
        __builtin_prefetch(kb + (hbase + kb0 + 32 + key) * DH + d0, 0, 1);
        __builtin_prefetch(vb + (hbase + kb0 + 32 + key) * DH + d0, 0, 1);
      }
      asm volatile("s_wait_asynccnt 0x0" ::: "memory");
    }
    __syncthreads();

    if (kb0 > qrow_base + 15) continue;        // block fully masked for this wave

    // S = Q K^T : B-fragments read K's natural [key][d] layout as B^T.
    v8f s0 = {}, s1 = {};
    s0 = __builtin_amdgcn_wmma_f32_16x16x32_f16(false, qf0, false, ld_frag_bT(&lK[0],            SK), (short)0, s0, false, false);
    s0 = __builtin_amdgcn_wmma_f32_16x16x32_f16(false, qf1, false, ld_frag_bT(&lK[32],           SK), (short)0, s0, false, false);
    s1 = __builtin_amdgcn_wmma_f32_16x16x32_f16(false, qf0, false, ld_frag_bT(&lK[16 * SK],      SK), (short)0, s1, false, false);
    s1 = __builtin_amdgcn_wmma_f32_16x16x32_f16(false, qf1, false, ld_frag_bT(&lK[16 * SK + 32], SK), (short)0, s1, false, false);

    _Float16* lPw = &lP[wv][0];
    float Alpha[8];
#pragma unroll
    for (int r = 0; r < 8; ++r) {
      const int qrow = qrow_base + r + hl * 8;
      float a = s0[r] * scl;
      float b = s1[r] * scl;
      if (kb0 + nn > qrow)      a = -__builtin_inff();
      if (kb0 + 16 + nn > qrow) b = -__builtin_inff();
      float mx = fmaxf(a, b);
#pragma unroll
      for (int off = 1; off < 16; off <<= 1) mx = fmaxf(mx, __shfl_xor(mx, off, 32));
      const float mnew  = fmaxf(Mrow[r], mx);
      const float alpha = __expf(Mrow[r] - mnew);
      const float pa = __expf(a - mnew);
      const float pb = __expf(b - mnew);
      float sum = pa + pb;
#pragma unroll
      for (int off = 1; off < 16; off <<= 1) sum += __shfl_xor(sum, off, 32);
      Lrow[r]  = Lrow[r] * alpha + sum;
      Mrow[r]  = mnew;
      Alpha[r] = alpha;
      lPw[(r + hl * 8) * SP + nn]      = (_Float16)pa;
      lPw[(r + hl * 8) * SP + 16 + nn] = (_Float16)pb;
    }
#pragma unroll
    for (int r = 0; r < 8; ++r) {
      o0[r] *= Alpha[r]; o1[r] *= Alpha[r]; o2[r] *= Alpha[r]; o3[r] *= Alpha[r];
    }

    // O += P V : A = staged P (same-wave LDS, HW keeps DS in order), B^T = lVT.
    const v16h pf = ld_frag_a(lPw, SP);
    o0 = __builtin_amdgcn_wmma_f32_16x16x32_f16(false, pf, false, ld_frag_bT(&lVT[ 0 * SV], SV), (short)0, o0, false, false);
    o1 = __builtin_amdgcn_wmma_f32_16x16x32_f16(false, pf, false, ld_frag_bT(&lVT[16 * SV], SV), (short)0, o1, false, false);
    o2 = __builtin_amdgcn_wmma_f32_16x16x32_f16(false, pf, false, ld_frag_bT(&lVT[32 * SV], SV), (short)0, o2, false, false);
    o3 = __builtin_amdgcn_wmma_f32_16x16x32_f16(false, pf, false, ld_frag_bT(&lVT[48 * SV], SV), (short)0, o3, false, false);
  }

  // out[t][h*64 + d] = O / l
#pragma unroll
  for (int r = 0; r < 8; ++r) {
    const int qrow  = qrow_base + r + hl * 8;
    const float inv = 1.0f / Lrow[r];
    float* op = out + (size_t)qrow * C_DIM + h * DH;
    op[nn]      = o0[r] * inv;
    op[16 + nn] = o1[r] * inv;
    op[32 + nn] = o2[r] * inv;
    op[48 + nn] = o3[r] * inv;
  }
}

extern "C" void kernel_launch(void* const* d_in, const int* in_sizes, int n_in,
                              void* d_out, int out_size, void* d_ws, size_t ws_size,
                              hipStream_t stream) {
  (void)in_sizes; (void)n_in; (void)out_size; (void)ws_size;
  const float* x = (const float*)d_in[0];       // [1,4096,768] fp32
  const float* w = (const float*)d_in[1];       // [768,2304]  fp32

  const size_t per = (size_t)NH * T_SEQ * DH;   // 3.1M halves each
  _Float16* qb = (_Float16*)d_ws;
  _Float16* kb = qb + per;
  _Float16* vb = kb + per;                      // total ~18.9 MB of d_ws

  dim3 g1(T_SEQ / 64, (3 * C_DIM) / 64);        // 64 x 36 tiles
  qkv_proj_wmma<<<g1, 256, 0, stream>>>(x, w, qb, kb, vb);

  dim3 g2(T_SEQ / 128, NH);                     // 32 x 12
  flash_attn_wmma<<<g2, 256, 0, stream>>>(qb, kb, vb, (float*)d_out);
}
